// EnhancedMultiHeadAttention_32676111188076
// MI455X (gfx1250) — matrix operand order repaired and checked by reading, not compiled, not run on hardware
//
#include <hip/hip_runtime.h>
#include <hip/hip_bf16.h>

// EnhancedMultiHeadAttention for MI455X (gfx1250), bf16 WMMA pipeline.
// B=32, C=512, L=512, H=8, DK=64, KS=7.

typedef __bf16 bf16;
typedef __attribute__((ext_vector_type(16))) __bf16 bf16x16;
typedef __attribute__((ext_vector_type(8)))  float  f32x8;

static __device__ inline f32x8 zero8() {
  f32x8 z;
#pragma unroll
  for (int i = 0; i < 8; ++i) z[i] = 0.0f;
  return z;
}

static __device__ inline f32x8 wmma_bf16(bf16x16 a, bf16x16 b, f32x8 c) {
  // D = A(16x32) * B(32x16) + C, f32 accumulate
  return __builtin_amdgcn_wmma_f32_16x16x32_bf16(
      false, a, false, b, (short)0, c, false, false);
}

// A-fragment (M=16 x K=32), source row-major [m][k], ld in elements.
// Lane l: m = l%16; K chunk = (l/16)*8 .. +7 and +16..+23  -> two 16B loads.
static __device__ inline bf16x16 load_frag_a(const bf16* base, int ld) {
  int lane = threadIdx.x & 31;
  const bf16* p = base + (size_t)(lane & 15) * ld + ((lane >> 4) << 3);
  union { bf16x16 v; uint4 u[2]; } t;
  t.u[0] = *reinterpret_cast<const uint4*>(p);
  t.u[1] = *reinterpret_cast<const uint4*>(p + 16);
  return t.v;
}

// B-fragment (K=32 x N=16), source is B^T row-major [n][k], ld in elements.
// Lane l: n = l%16; K chunk = (l/16)*8 .. +7 and +16..+23  -> two 16B loads.
static __device__ inline bf16x16 load_frag_b(const bf16* baseT, int ld) {
  int lane = threadIdx.x & 31;
  const bf16* p = baseT + (size_t)(lane & 15) * ld + ((lane >> 4) << 3);
  union { bf16x16 v; uint4 u[2]; } t;
  t.u[0] = *reinterpret_cast<const uint4*>(p);
  t.u[1] = *reinterpret_cast<const uint4*>(p + 16);
  return t.v;
}

// 16-lane-group reductions (groups = lanes 0..15 / 16..31, matching the
// C/D fragment row mapping M = r + 8*(lane/16)).
static __device__ inline float redmax16(float v) {
#pragma unroll
  for (int m = 1; m < 16; m <<= 1) v = fmaxf(v, __shfl_xor(v, m, 32));
  return v;
}
static __device__ inline float redsum16(float v) {
#pragma unroll
  for (int m = 1; m < 16; m <<= 1) v += __shfl_xor(v, m, 32);
  return v;
}

// ---------------------------------------------------------------- convert
__global__ void cvt_f32_bf16(const float* __restrict__ s, bf16* __restrict__ d, int n) {
  int i = blockIdx.x * 256 + threadIdx.x;
  if (i < n) d[i] = (bf16)s[i];
}

// ------------------------------------------------- depthwise conv (+pos)
// x:(B,C,L) f32 -> y_t:(B,L,C) bf16 ;  in = x + pos_bias[l,c] (if pos)
__global__ __launch_bounds__(512) void dwconv_kernel(
    const float* __restrict__ x, const float* __restrict__ pos,
    const float* __restrict__ dww, const float* __restrict__ dwb,
    bf16* __restrict__ yt) {
  int c = threadIdx.x;             // 0..511
  int l = blockIdx.x & 511;
  int b = blockIdx.x >> 9;
  const float* xb = x + ((size_t)b * 512 + c) * 512;
  float acc = dwb[c];
#pragma unroll
  for (int kk = 0; kk < 7; ++kk) {
    int ll = l + kk - 3;
    if (0 <= ll && ll < 512) {
      float v = xb[ll];
      if (pos) v += pos[(size_t)ll * 512 + c];
      acc += v * dww[c * 7 + kk];
    }
  }
  yt[((size_t)b * 512 + l) * 512 + c] = (bf16)acc;
}

// -------------------------------------------------- pointwise GEMM (q/k/v)
// Z[o,l] = sum_c W[o,c] * Yt[l,c] + bias[o]
// mode 0: out (B,H,L,DK)  [q,k: transposed within head]
// mode 1: out (B,H,DK,L)  [v]
// grid (4,4,32) block 256 (8 waves: 2M x 4N), block tile 128x128, wave 64x32.
__global__ __launch_bounds__(256) void qkv_gemm(
    const bf16* __restrict__ W, const bf16* __restrict__ Yt,
    const float* __restrict__ bias, bf16* __restrict__ out, int mode) {
  int b = blockIdx.z;
  int wave = threadIdx.x >> 5, lane = threadIdx.x & 31;
  int wm = wave >> 2, wn = wave & 3;
  int o0 = blockIdx.x * 128 + wm * 64;
  int l0 = blockIdx.y * 128 + wn * 32;
  const bf16* Yb = Yt + (size_t)b * 512 * 512;

  f32x8 acc[4][2];
#pragma unroll
  for (int mt = 0; mt < 4; ++mt)
#pragma unroll
    for (int nt = 0; nt < 2; ++nt) acc[mt][nt] = zero8();

  for (int k0 = 0; k0 < 512; k0 += 32) {
    bf16x16 af[4], bfr[2];
#pragma unroll
    for (int mt = 0; mt < 4; ++mt)
      af[mt] = load_frag_a(W + (size_t)(o0 + mt * 16) * 512 + k0, 512);
#pragma unroll
    for (int nt = 0; nt < 2; ++nt)
      bfr[nt] = load_frag_b(Yb + (size_t)(l0 + nt * 16) * 512 + k0, 512);
#pragma unroll
    for (int mt = 0; mt < 4; ++mt)
#pragma unroll
      for (int nt = 0; nt < 2; ++nt)
        acc[mt][nt] = wmma_bf16(af[mt], bfr[nt], acc[mt][nt]);
  }

  int n = lane & 15, hi = lane >> 4;
#pragma unroll
  for (int mt = 0; mt < 4; ++mt) {
#pragma unroll
    for (int nt = 0; nt < 2; ++nt) {
      int l = l0 + nt * 16 + n;
      int obase = o0 + mt * 16 + 8 * hi;      // o = obase + r, r=0..7 (one head)
      union { bf16 e[8]; uint4 u; } pk;
#pragma unroll
      for (int r = 0; r < 8; ++r)
        pk.e[r] = (bf16)(acc[mt][nt][r] + bias[obase + r]);
      if (mode == 0) {
        // (B,H,L,DK): contiguous in o within head -> one 16B store
        bf16* dst = out + ((size_t)(b * 8 + (obase >> 6)) * 512 + l) * 64 + (obase & 63);
        *reinterpret_cast<uint4*>(dst) = pk.u;
      } else {
        // (B,H,DK,L): stride-512 scalar stores
#pragma unroll
        for (int r = 0; r < 8; ++r) {
          int o = obase + r;
          out[((size_t)(b * 8 + (o >> 6)) * 64 + (o & 63)) * 512 + l] = pk.e[r];
        }
      }
    }
  }
}

// -------------------------------------------------------- flash attention
// qt,kt:(B,H,L,DK) bf16 ; vdl:(B,H,DK,L) bf16 ; ao:(B,C,L) bf16
// grid 1024 = B*H*4, block 256 = 8 waves, each wave owns 16 rows of i.
__global__ __launch_bounds__(256) void attn_kernel(
    const bf16* __restrict__ qt, const bf16* __restrict__ kt,
    const bf16* __restrict__ vdl, bf16* __restrict__ ao) {
  __shared__ __align__(16) bf16 pbuf[8][16 * 32];  // per-wave P staging (1KB)
  int wave = threadIdx.x >> 5, lane = threadIdx.x & 31;
  int n = lane & 15, hi = lane >> 4;
  int bh = blockIdx.x >> 2;                 // b*8 + h
  int i0 = (blockIdx.x & 3) * 128 + wave * 16;

  const bf16* qbase = qt + ((size_t)bh * 512 + i0) * 64;
  const bf16* kbase = kt + (size_t)bh * 512 * 64;
  const bf16* vbase = vdl + (size_t)bh * 64 * 512;
  bf16* pb = pbuf[wave];

  bf16x16 aq0 = load_frag_a(qbase, 64);        // K = d 0..31
  bf16x16 aq1 = load_frag_a(qbase + 32, 64);   // K = d 32..63

  float m[8], lsum[8];
  f32x8 oacc[4];
#pragma unroll
  for (int r = 0; r < 8; ++r) { m[r] = -1e30f; lsum[r] = 0.0f; }
#pragma unroll
  for (int t = 0; t < 4; ++t) oacc[t] = zero8();

  for (int j0 = 0; j0 < 512; j0 += 32) {
    // scores tiles: rows i (16), cols j0..j0+15 and j0+16..j0+31, K=d 64
    f32x8 s0 = zero8(), s1 = zero8();
    s0 = wmma_bf16(aq0, load_frag_b(kbase + (size_t)j0 * 64, 64), s0);
    s0 = wmma_bf16(aq1, load_frag_b(kbase + (size_t)j0 * 64 + 32, 64), s0);
    s1 = wmma_bf16(aq0, load_frag_b(kbase + (size_t)(j0 + 16) * 64, 64), s1);
    s1 = wmma_bf16(aq1, load_frag_b(kbase + (size_t)(j0 + 16) * 64 + 32, 64), s1);

    float scale[8];
#pragma unroll
    for (int r = 0; r < 8; ++r) {
      float a = s0[r] * 0.125f, b2 = s1[r] * 0.125f;   // 1/sqrt(64)
      float nm = fmaxf(m[r], redmax16(fmaxf(a, b2)));
      scale[r] = __expf(m[r] - nm);
      m[r] = nm;
      float p0 = __expf(a - nm), p1 = __expf(b2 - nm);
      lsum[r] = lsum[r] * scale[r] + redsum16(p0 + p1);
      int M = r + 8 * hi;                    // transpose P through LDS
      pb[M * 32 + n]      = (bf16)p0;
      pb[M * 32 + 16 + n] = (bf16)p1;
    }
#pragma unroll
    for (int t = 0; t < 4; ++t)
#pragma unroll
      for (int r = 0; r < 8; ++r) oacc[t][r] *= scale[r];

    asm volatile("s_wait_dscnt 0" ::: "memory");  // P stores visible to A-frag loads
    bf16x16 pa = load_frag_a(pb, 32);             // P as A (16 x K=32)
#pragma unroll
    for (int t = 0; t < 4; ++t) {
      // B^T = v_dl rows d = t*16.., cols j
      bf16x16 vb = load_frag_b(vbase + (size_t)(t * 16) * 512 + j0, 512);
      oacc[t] = wmma_bf16(pa, vb, oacc[t]);
    }
  }

  float rl[8];
#pragma unroll
  for (int r = 0; r < 8; ++r) rl[r] = 1.0f / lsum[r];

  int b = bh >> 3, h = bh & 7;
#pragma unroll
  for (int t = 0; t < 4; ++t) {
    union { bf16 e[8]; uint4 u; } pk;
#pragma unroll
    for (int r = 0; r < 8; ++r) pk.e[r] = (bf16)(oacc[t][r] * rl[r]);
    // D tile (M=i, N=d) stored transposed into AO[b, c=h*64+d, l=i]
    bf16* dst = ao + ((size_t)(b * 512 + h * 64 + t * 16 + n)) * 512 + i0 + 8 * hi;
    *reinterpret_cast<uint4*>(dst) = pk.u;
  }
}

// ----------------------------------------------------- final projection
// res[b,c,o] = sum_l AO[b,c,l] * Wp[o,l] + pb[o]   (M=o, N=c, K=l)
__global__ __launch_bounds__(256) void proj_gemm(
    const bf16* __restrict__ Wp, const bf16* __restrict__ AO,
    const float* __restrict__ pbias, float* __restrict__ res) {
  int b = blockIdx.z;
  int wave = threadIdx.x >> 5, lane = threadIdx.x & 31;
  int wm = wave >> 2, wn = wave & 3;
  int o0 = blockIdx.x * 128 + wm * 64;
  int c0 = blockIdx.y * 128 + wn * 32;
  const bf16* Ab = AO + (size_t)b * 512 * 512;

  f32x8 acc[4][2];
#pragma unroll
  for (int mt = 0; mt < 4; ++mt)
#pragma unroll
    for (int nt = 0; nt < 2; ++nt) acc[mt][nt] = zero8();

  for (int k0 = 0; k0 < 512; k0 += 32) {
    bf16x16 af[4], bfr[2];
#pragma unroll
    for (int mt = 0; mt < 4; ++mt)
      af[mt] = load_frag_a(Wp + (size_t)(o0 + mt * 16) * 512 + k0, 512);
#pragma unroll
    for (int nt = 0; nt < 2; ++nt)
      bfr[nt] = load_frag_b(Ab + (size_t)(c0 + nt * 16) * 512 + k0, 512);
#pragma unroll
    for (int mt = 0; mt < 4; ++mt)
#pragma unroll
      for (int nt = 0; nt < 2; ++nt)
        acc[mt][nt] = wmma_bf16(af[mt], bfr[nt], acc[mt][nt]);
  }

  int n = lane & 15, hi = lane >> 4;
#pragma unroll
  for (int mt = 0; mt < 4; ++mt) {
#pragma unroll
    for (int nt = 0; nt < 2; ++nt) {
      int c = c0 + nt * 16 + n;
      int obase = o0 + mt * 16 + 8 * hi;
      float* dst = res + (size_t)b * 512 * 512 + (size_t)c * 512 + obase;
      float4 v0, v1;
#pragma unroll
      for (int r = 0; r < 4; ++r) {
        (&v0.x)[r] = acc[mt][nt][r] + pbias[obase + r];
        (&v1.x)[r] = acc[mt][nt][r + 4] + pbias[obase + r + 4];
      }
      *reinterpret_cast<float4*>(dst) = v0;
      *reinterpret_cast<float4*>(dst + 4) = v1;
    }
  }
}

// ------------------------------------------------------------------ host
extern "C" void kernel_launch(void* const* d_in, const int* in_sizes, int n_in,
                              void* d_out, int out_size, void* d_ws, size_t ws_size,
                              hipStream_t stream) {
  const float* query  = (const float*)d_in[0];
  const float* key_   = (const float*)d_in[1];
  const float* value  = (const float*)d_in[2];
  const float* pos    = (const float*)d_in[3];
  const float* proj_w = (const float*)d_in[4];
  const float* proj_b = (const float*)d_in[5];
  const float* q_dw_w = (const float*)d_in[6];
  const float* q_dw_b = (const float*)d_in[7];
  const float* q_pw_w = (const float*)d_in[8];
  const float* q_pw_b = (const float*)d_in[9];
  const float* k_dw_w = (const float*)d_in[10];
  const float* k_dw_b = (const float*)d_in[11];
  const float* k_pw_w = (const float*)d_in[12];
  const float* k_pw_b = (const float*)d_in[13];
  const float* v_dw_w = (const float*)d_in[14];
  const float* v_dw_b = (const float*)d_in[15];
  const float* v_pw_w = (const float*)d_in[16];
  const float* v_pw_b = (const float*)d_in[17];

  const size_t NBC = (size_t)32 * 512 * 512;  // per-tensor elements
  bf16* ws   = (bf16*)d_ws;
  bf16* qt   = ws;                 // (B,H,L,DK)
  bf16* kt   = qt + NBC;           // (B,H,L,DK)
  bf16* vdl  = kt + NBC;           // (B,H,DK,L)
  bf16* ybuf = vdl + NBC;          // dsconv out (B,L,C); reused as AO (B,C,L)
  bf16* wq   = ybuf + NBC;
  bf16* wk   = wq + 512 * 512;
  bf16* wv   = wk + 512 * 512;
  bf16* wp   = wv + 512 * 512;
  float* out = (float*)d_out;

  cvt_f32_bf16<<<1024, 256, 0, stream>>>(q_pw_w, wq, 512 * 512);
  cvt_f32_bf16<<<1024, 256, 0, stream>>>(k_pw_w, wk, 512 * 512);
  cvt_f32_bf16<<<1024, 256, 0, stream>>>(v_pw_w, wv, 512 * 512);
  cvt_f32_bf16<<<1024, 256, 0, stream>>>(proj_w, wp, 512 * 512);

  dim3 gemm_grid(4, 4, 32);

  dwconv_kernel<<<16384, 512, 0, stream>>>(query, pos, q_dw_w, q_dw_b, ybuf);
  qkv_gemm<<<gemm_grid, 256, 0, stream>>>(wq, ybuf, q_pw_b, qt, 0);

  dwconv_kernel<<<16384, 512, 0, stream>>>(key_, pos, k_dw_w, k_dw_b, ybuf);
  qkv_gemm<<<gemm_grid, 256, 0, stream>>>(wk, ybuf, k_pw_b, kt, 0);

  dwconv_kernel<<<16384, 512, 0, stream>>>(value, nullptr, v_dw_w, v_dw_b, ybuf);
  qkv_gemm<<<gemm_grid, 256, 0, stream>>>(wv, ybuf, v_pw_b, vdl, 1);

  attn_kernel<<<1024, 256, 0, stream>>>(qt, kt, vdl, ybuf);

  proj_gemm<<<gemm_grid, 256, 0, stream>>>(wp, ybuf, proj_b, out);
}
